// ConvolutionNN_26517128085983
// MI455X (gfx1250) — matrix operand order repaired and checked by reading, not compiled, not run on hardware
//
#include <hip/hip_runtime.h>

// ---------------------------------------------------------------------------
// LeNet-ish forward for MI455X (gfx1250, wave32).
// 5 kernels on one stream:
//   k_stats1  : conv1 per image -> per-channel sum/sumsq block partials
//   k_reduce  : fixed-order reduce -> BN affine coeffs (a,b) per channel
//   k_stats2  : conv1+BN1+relu+pool+conv2 -> per-channel sum/sumsq partials
//   k_reduce  : -> BN2 affine coeffs
//   k_final   : full recompute -> 64 f16 features/img in LDS -> FC chain via
//               v_wmma_f32_16x16x32_f16 (A/B fragments built per ISA layout)
// Deterministic: no atomics, reductions in fixed order.
// Workspace usage: ~90 KB of fp32 partials/coeffs only.
// ---------------------------------------------------------------------------

typedef __attribute__((ext_vector_type(16))) _Float16 v16h;
typedef __attribute__((ext_vector_type(8)))  float    v8f;

#define EPS  1e-5f
#define TPB  256
#define NB   512           // 512 * 256 = 131072 images

#define WMMA_F16(A, B, C) \
  __builtin_amdgcn_wmma_f32_16x16x32_f16(false, (A), false, (B), (short)0, (C), false, false)

static __device__ __forceinline__ float wave_sum(float v) {
#pragma unroll
  for (int o = 16; o > 0; o >>= 1) v += __shfl_xor(v, o, 32);
  return v;
}

// conv1: 1->6, k=3, pad=2, in 8x8 -> out 10x10 (row i, all 10 cols)
static __device__ __forceinline__ void conv1_row(const float* xr, const float* wc,
                                                 float bias, int i, float* out) {
#pragma unroll
  for (int j = 0; j < 10; ++j) {
    float acc = bias;
#pragma unroll
    for (int u = 0; u < 3; ++u) {
      int y = i + u - 2;
      if (y < 0 || y > 7) continue;
#pragma unroll
      for (int v = 0; v < 3; ++v) {
        int xx = j + v - 2;
        if (xx < 0 || xx > 7) continue;
        acc = fmaf(wc[u * 3 + v], xr[y * 8 + xx], acc);
      }
    }
    out[j] = acc;
  }
}

// conv1 -> BN1 (affine a,b) -> relu -> 2x2 meanpool : pooled[6][5][5]
static __device__ __forceinline__ void conv1_bn_pool(const float* xr,
    const float* __restrict__ w1, const float* __restrict__ b1,
    const float* __restrict__ ab1, float* pooled) {
#pragma unroll
  for (int c = 0; c < 6; ++c) {
    float wc[9];
#pragma unroll
    for (int k = 0; k < 9; ++k) wc[k] = w1[c * 9 + k];
    float bias = b1[c];
    float a = ab1[c], bb = ab1[6 + c];
#pragma unroll
    for (int pr = 0; pr < 5; ++pr) {
      float r0[10], r1[10];
      conv1_row(xr, wc, bias, 2 * pr, r0);
      conv1_row(xr, wc, bias, 2 * pr + 1, r1);
#pragma unroll
      for (int pc = 0; pc < 5; ++pc) {
        float s = fmaxf(fmaf(a, r0[2 * pc],     bb), 0.f)
                + fmaxf(fmaf(a, r0[2 * pc + 1], bb), 0.f)
                + fmaxf(fmaf(a, r1[2 * pc],     bb), 0.f)
                + fmaxf(fmaf(a, r1[2 * pc + 1], bb), 0.f);
        pooled[c * 25 + pr * 5 + pc] = 0.25f * s;
      }
    }
  }
}

// conv2 (6->16, k=2) at compile-time position (i,j), runtime oc weights wv[24]
static __device__ __forceinline__ float conv2_at(const float* pooled, const float* wv,
                                                 float bo, int i, int j) {
  float acc = bo;
#pragma unroll
  for (int c = 0; c < 6; ++c)
#pragma unroll
    for (int u = 0; u < 2; ++u)
#pragma unroll
      for (int v = 0; v < 2; ++v)
        acc = fmaf(wv[c * 4 + u * 2 + v], pooled[c * 25 + (i + u) * 5 + (j + v)], acc);
  return acc;
}

static __device__ __forceinline__ void load_img(const float* __restrict__ x, int img,
                                                float* xr) {
  const float4* p = (const float4*)(x + (size_t)img * 64);
#pragma unroll
  for (int i = 0; i < 16; ++i) ((float4*)xr)[i] = p[i];
}

// -------------------------- kernel 1: conv1 stats ---------------------------
__global__ __launch_bounds__(TPB) void k_stats1(const float* __restrict__ x,
    const float* __restrict__ w1, const float* __restrict__ b1,
    float* __restrict__ partial) {
  __shared__ float lred[8];
  const int tid = threadIdx.x, lane = tid & 31, wid = tid >> 5;
  const int img = blockIdx.x * TPB + tid;
  float xr[64];
  load_img(x, img, xr);

  float s[6], ss[6];
#pragma unroll
  for (int c = 0; c < 6; ++c) {
    float wc[9];
#pragma unroll
    for (int k = 0; k < 9; ++k) wc[k] = w1[c * 9 + k];
    float bias = b1[c];
    float sc = 0.f, ssc = 0.f;
#pragma unroll
    for (int i = 0; i < 10; ++i) {
      float r[10];
      conv1_row(xr, wc, bias, i, r);
#pragma unroll
      for (int j = 0; j < 10; ++j) { sc += r[j]; ssc = fmaf(r[j], r[j], ssc); }
    }
    s[c] = sc; ss[c] = ssc;
  }
#pragma unroll
  for (int q = 0; q < 12; ++q) {
    float v = (q < 6) ? s[q] : ss[q - 6];
    v = wave_sum(v);
    if (lane == 0) lred[wid] = v;
    __syncthreads();
    if (tid == 0) {
      float t = 0.f;
#pragma unroll
      for (int w = 0; w < 8; ++w) t += lred[w];
      partial[blockIdx.x * 12 + q] = t;
    }
    __syncthreads();
  }
}

// --------------------- fixed-order reduce -> BN coeffs ----------------------
__global__ void k_reduce(const float* __restrict__ partial, const float* __restrict__ g,
                         const float* __restrict__ be, float* __restrict__ ab,
                         int nblocks, int nchan, float inv_n) {
  int c = threadIdx.x;
  if (c >= nchan) return;
  float s = 0.f, ss = 0.f;
  for (int b = 0; b < nblocks; ++b) {
    s  += partial[b * 2 * nchan + c];
    ss += partial[b * 2 * nchan + nchan + c];
  }
  float mean = s * inv_n;
  float var  = ss * inv_n - mean * mean;
  float istd = rsqrtf(var + EPS);
  float a = g[c] * istd;
  ab[c] = a;
  ab[nchan + c] = be[c] - a * mean;
}

// -------------------------- kernel 3: conv2 stats ---------------------------
__global__ __launch_bounds__(TPB) void k_stats2(const float* __restrict__ x,
    const float* __restrict__ w1, const float* __restrict__ b1,
    const float* __restrict__ ab1, const float* __restrict__ w2,
    const float* __restrict__ b2, float* __restrict__ partial) {
  __shared__ float lred[8];
  const int tid = threadIdx.x, lane = tid & 31, wid = tid >> 5;
  const int img = blockIdx.x * TPB + tid;
  float xr[64];
  load_img(x, img, xr);
  float pooled[150];
  conv1_bn_pool(xr, w1, b1, ab1, pooled);

  for (int oc = 0; oc < 16; ++oc) {           // uniform runtime loop
    float wv[24];
#pragma unroll
    for (int k = 0; k < 24; ++k) wv[k] = w2[oc * 24 + k];
    float bo = b2[oc];
    float sc = 0.f, ssc = 0.f;
#pragma unroll
    for (int i = 0; i < 4; ++i)
#pragma unroll
      for (int j = 0; j < 4; ++j) {
        float o = conv2_at(pooled, wv, bo, i, j);
        sc += o; ssc = fmaf(o, o, ssc);
      }
    sc = wave_sum(sc); ssc = wave_sum(ssc);
    if (lane == 0) lred[wid] = sc;
    __syncthreads();
    if (tid == 0) {
      float t = 0.f;
#pragma unroll
      for (int w = 0; w < 8; ++w) t += lred[w];
      partial[blockIdx.x * 32 + oc] = t;
    }
    __syncthreads();
    if (lane == 0) lred[wid] = ssc;
    __syncthreads();
    if (tid == 0) {
      float t = 0.f;
#pragma unroll
      for (int w = 0; w < 8; ++w) t += lred[w];
      partial[blockIdx.x * 32 + 16 + oc] = t;
    }
    __syncthreads();
  }
}

// ---------------- kernel 5: fused recompute + WMMA FC chain -----------------
__global__ __launch_bounds__(TPB) void k_final(
    const float* __restrict__ x,  const float* __restrict__ w1, const float* __restrict__ b1,
    const float* __restrict__ ab1, const float* __restrict__ w2, const float* __restrict__ b2,
    const float* __restrict__ ab2,
    const float* __restrict__ fw1, const float* __restrict__ fb1,
    const float* __restrict__ fw2, const float* __restrict__ fb2,
    const float* __restrict__ fw3, const float* __restrict__ fb3,
    float* __restrict__ out) {
  __shared__ _Float16 feat[TPB][72];        // 64 feats/img, padded stride
  __shared__ _Float16 h1buf[8][16][34];     // per-wave FC1 output staging (16x32)
  __shared__ _Float16 h2buf[8][16][18];     // per-wave FC2 output staging (16x16)

  const int tid = threadIdx.x, lane = tid & 31, wid = tid >> 5;
  const int blockBase = blockIdx.x * TPB;
  const int img = blockBase + tid;

  // ---- per-image feature extraction (VALU) ----
  {
    float xr[64];
    load_img(x, img, xr);
    float pooled[150];
    conv1_bn_pool(xr, w1, b1, ab1, pooled);
    for (int oc = 0; oc < 16; ++oc) {       // uniform runtime loop
      float wv[24];
#pragma unroll
      for (int k = 0; k < 24; ++k) wv[k] = w2[oc * 24 + k];
      float bo = b2[oc], a2 = ab2[oc], c2 = ab2[16 + oc];
#pragma unroll
      for (int p = 0; p < 4; ++p) {
        int py = p >> 1, px = p & 1;
        float s = 0.f;
#pragma unroll
        for (int dy = 0; dy < 2; ++dy)
#pragma unroll
          for (int dx = 0; dx < 2; ++dx) {
            float o = conv2_at(pooled, wv, bo, 2 * py + dy, 2 * px + dx);
            s += fmaxf(fmaf(a2, o, c2), 0.f);
          }
        feat[tid][oc * 4 + p] = (_Float16)(0.25f * s);
      }
    }
  }
  __syncthreads();

  // ---- FC chain via WMMA (wave handles 32 images as two 16-row tiles) ----
  const int ncol = lane & 15;

  // B fragments per ISA layout: element e -> K = (lane<16 ? 0 : 8) + (e<8 ? e : e+8)
  v16h bf1[2][2], bf2, bf3;
#pragma unroll
  for (int nt = 0; nt < 2; ++nt) {
    int n = ncol + nt * 16;
#pragma unroll
    for (int kc = 0; kc < 2; ++kc)
#pragma unroll
      for (int e = 0; e < 16; ++e) {
        int k = ((lane < 16) ? 0 : 8) + ((e < 8) ? e : e + 8) + kc * 32;
        bf1[nt][kc][e] = (n < 30) ? (_Float16)fw1[n * 64 + k] : (_Float16)0.f;
      }
  }
#pragma unroll
  for (int e = 0; e < 16; ++e) {
    int k = ((lane < 16) ? 0 : 8) + ((e < 8) ? e : e + 8);
    bf2[e] = (ncol < 15 && k < 30) ? (_Float16)fw2[ncol * 30 + k] : (_Float16)0.f;
    bf3[e] = (ncol < 10 && k < 15) ? (_Float16)fw3[ncol * 15 + k] : (_Float16)0.f;
  }
  const float fb1_lo = fb1[ncol];
  const float fb1_hi = (ncol + 16 < 30) ? fb1[ncol + 16] : 0.f;
  const float fb2_n  = (ncol < 15) ? fb2[ncol] : 0.f;
  const float fb3_n  = (ncol < 10) ? fb3[ncol] : 0.f;
  const int kbaseA = (lane < 16) ? 0 : 8;   // A layout: K = kbaseA + (e<8 ? e : e+8)
  const int mhi    = (lane >= 16) ? 8 : 0;  // D layout: row = vgpr + mhi

  for (int t = 0; t < 2; ++t) {             // uniform tile loop
    const int arow = wid * 32 + t * 16 + ncol;   // local image row for A
    // FC1: A = feat[16 x 64] f16, two K-chunks
    v16h a0, a1;
#pragma unroll
    for (int e = 0; e < 16; ++e) {
      int k = kbaseA + ((e < 8) ? e : e + 8);
      a0[e] = feat[arow][k];
      a1[e] = feat[arow][k + 32];
    }
    v8f acc0 = {}, acc1 = {};
    acc0 = WMMA_F16(a0, bf1[0][0], acc0);
    acc0 = WMMA_F16(a1, bf1[0][1], acc0);
    acc1 = WMMA_F16(a0, bf1[1][0], acc1);
    acc1 = WMMA_F16(a1, bf1[1][1], acc1);
#pragma unroll
    for (int r = 0; r < 8; ++r) {
      int M = r + mhi;
      h1buf[wid][M][ncol]      = (_Float16)fmaxf(acc0[r] + fb1_lo, 0.f);
      h1buf[wid][M][ncol + 16] = (_Float16)fmaxf(acc1[r] + fb1_hi, 0.f);
    }
    __syncthreads();

    // FC2: A = h1 [16 x 32] (cols 30,31 are exact zeros)
    v16h a2v;
#pragma unroll
    for (int e = 0; e < 16; ++e) {
      int k = kbaseA + ((e < 8) ? e : e + 8);
      a2v[e] = h1buf[wid][ncol][k];
    }
    v8f acc2 = {};
    acc2 = WMMA_F16(a2v, bf2, acc2);
#pragma unroll
    for (int r = 0; r < 8; ++r) {
      int M = r + mhi;
      h2buf[wid][M][ncol] = (_Float16)fmaxf(acc2[r] + fb2_n, 0.f);
    }
    __syncthreads();

    // FC3: A = h2 [16 x 16] (col 15 zero), K padded to 32 with zeros
    v16h a3v;
#pragma unroll
    for (int e = 0; e < 16; ++e) {
      int k = kbaseA + ((e < 8) ? e : e + 8);
      a3v[e] = (k < 16) ? h2buf[wid][ncol][k] : (_Float16)0.f;
    }
    v8f acc3 = {};
    acc3 = WMMA_F16(a3v, bf3, acc3);
#pragma unroll
    for (int r = 0; r < 8; ++r) {
      int M = r + mhi;
      if (ncol < 10)
        out[(size_t)(blockBase + wid * 32 + t * 16 + M) * 10 + ncol] = acc3[r] + fb3_n;
    }
    __syncthreads();
  }
}

// ---------------------------------------------------------------------------
extern "C" void kernel_launch(void* const* d_in, const int* in_sizes, int n_in,
                              void* d_out, int out_size, void* d_ws, size_t ws_size,
                              hipStream_t stream) {
  (void)in_sizes; (void)n_in; (void)out_size; (void)ws_size;
  const float* x   = (const float*)d_in[0];
  const float* w1  = (const float*)d_in[1];
  const float* b1  = (const float*)d_in[2];
  const float* g1  = (const float*)d_in[3];
  const float* be1 = (const float*)d_in[4];
  const float* w2  = (const float*)d_in[5];
  const float* b2  = (const float*)d_in[6];
  const float* g2  = (const float*)d_in[7];
  const float* be2 = (const float*)d_in[8];
  const float* fw1 = (const float*)d_in[9];
  const float* fb1 = (const float*)d_in[10];
  const float* fw2 = (const float*)d_in[11];
  const float* fb2 = (const float*)d_in[12];
  const float* fw3 = (const float*)d_in[13];
  const float* fb3 = (const float*)d_in[14];
  float* out = (float*)d_out;

  float* ws = (float*)d_ws;
  float* partial1 = ws;                 // 512*12 = 6144
  float* ab1      = ws + 6144;          // 12
  float* partial2 = ws + 6208;          // 512*32 = 16384
  float* ab2      = ws + 6208 + 16384;  // 32   (total < 23k floats ~ 91 KB)

  const float invN1 = 1.f / (131072.f * 100.f);
  const float invN2 = 1.f / (131072.f * 16.f);

  k_stats1<<<NB, TPB, 0, stream>>>(x, w1, b1, partial1);
  k_reduce<<<1, 32, 0, stream>>>(partial1, g1, be1, ab1, NB, 6,  invN1);
  k_stats2<<<NB, TPB, 0, stream>>>(x, w1, b1, ab1, w2, b2, partial2);
  k_reduce<<<1, 32, 0, stream>>>(partial2, g2, be2, ab2, NB, 16, invN2);
  k_final <<<NB, TPB, 0, stream>>>(x, w1, b1, ab1, w2, b2, ab2,
                                   fw1, fb1, fw2, fb2, fw3, fb3, out);
}
